// BatchProgramClassifier_42786464202770
// MI455X (gfx1250) — matrix operand order repaired and checked by reading, not compiled, not run on hardware
//
#include <hip/hip_runtime.h>
#include <hip/hip_bf16.h>
#include <math.h>

// ---------------------------------------------------------------------------
// Problem constants (from reference)
// ---------------------------------------------------------------------------
#define B_      64
#define L_      64
#define NPT_    127     // nodes per tree (full binary tree, depth 7)
#define D_      128
#define VOCAB_  30000
#define LABELS_ 30

typedef __bf16 bf16;
typedef __bf16 v16bf __attribute__((ext_vector_type(16)));
typedef float  v8f   __attribute__((ext_vector_type(8)));

// Workspace layout (bytes):
//   enc   : [0,           1048576)   B*L x 128 bf16
//   Wts   : [1048576,     1222656)   5 images of 128x136 bf16 (Wc,Wq,Wk,Wv,Wo)^T
//   embbf : [1222656,     8902656)   30000x128 bf16
#define WS_ENC_OFF   0
#define WS_WT_OFF    1048576
#define WT_IMG_ELEMS (128 * 136)
#define WT_IMG_BYTES (WT_IMG_ELEMS * 2)
#define WS_EMB_OFF   1222656

// ---------------------------------------------------------------------------
// Async global->LDS staging (gfx1250 GLOBAL_LOAD_ASYNC_TO_LDS_B128, ASYNCcnt)
// with a b128 load/store fallback if the builtins are not declared.
// Builtin signature (from hipcc diagnostic): param 0 is a vector-of-4-int
// pointer in the global address space (non-const); dest is LDS (as(3)).
// ---------------------------------------------------------------------------
#if defined(__AMDGCN__) && __has_builtin(__builtin_amdgcn_global_load_async_to_lds_b128) && __has_builtin(__builtin_amdgcn_s_wait_asynccnt)
#define USE_ASYNC_LDS 1
#else
#define USE_ASYNC_LDS 0
#endif

typedef int vsi4 __attribute__((vector_size(16)));

__device__ inline void copy16_g2l(void* lds_dst, const void* g_src) {
#if USE_ASYNC_LDS
  __builtin_amdgcn_global_load_async_to_lds_b128(
      (__attribute__((address_space(1))) vsi4*)g_src,
      (__attribute__((address_space(3))) vsi4*)lds_dst, 0, 0);
#else
  *(uint4*)lds_dst = *(const uint4*)g_src;
#endif
}

__device__ inline void async_fence() {
#if USE_ASYNC_LDS
  __builtin_amdgcn_s_wait_asynccnt(0);
#endif
}

// Flat copy of `bytes` (multiple of 16) from global to LDS, 256 threads.
__device__ inline void copy_flat_g2l(void* lds_dst, const void* g_src,
                                     int bytes, int tid) {
  for (int o = tid * 16; o < bytes; o += 256 * 16)
    copy16_g2l((char*)lds_dst + o, (const char*)g_src + o);
}

// ---------------------------------------------------------------------------
// WMMA helpers (gfx1250, wave32).  Layouts per ISA 7.12.2:
// A frag (16x32 bf16): lane L row M=L&15; half-wave K offsets {0,8};
//   regs 0-3 = K base..base+7, regs 4-7 = K base+16..base+23
//   => two 16B loads at +0 / +16 elements.
// B frag (32x16 bf16) from N-major K-contiguous image Bt[n][k]:
//   lane L col N=L&15; same K layout as A (half-wave K offsets {0,8})
//   => two 16B loads at +0 / +16 elements.
// C/D (16x16 f32): reg j, lane L -> M = j + 8*(L>=16), N = L&15.
// ---------------------------------------------------------------------------
__device__ inline v16bf load_frag_a(const bf16* base, int ld, int lane) {
  const bf16* p = base + (lane & 15) * ld + ((lane >> 4) << 3);
  union { v16bf v; uint4 q[2]; } u;
  u.q[0] = *(const uint4*)(p);
  u.q[1] = *(const uint4*)(p + 16);
  return u.v;
}

__device__ inline v16bf load_frag_b(const bf16* base, int ld, int lane) {
  const bf16* p = base + (lane & 15) * ld + ((lane >> 4) << 3);
  union { v16bf v; uint4 q[2]; } u;
  u.q[0] = *(const uint4*)(p);
  u.q[1] = *(const uint4*)(p + 16);
  return u.v;
}

__device__ inline v8f wmma_bf16(v16bf a, v16bf b, v8f c) {
  return __builtin_amdgcn_wmma_f32_16x16x32_bf16(false, a, false, b,
                                                 (short)0, c, false, false);
}

__device__ inline v8f zero_v8f() {
  v8f z;
  #pragma unroll
  for (int i = 0; i < 8; ++i) z[i] = 0.f;
  return z;
}

// ---------------------------------------------------------------------------
// Kernel 0: one-time precondition.
//  - emb (f32) -> embbf (bf16), flat
//  - {Wc,Wq,Wk,Wv,Wo} (128x128 f32 row-major W[k][n]) -> bf16 transposed
//    LDS-images WT[n*136+k], ready for byte-exact async staging.
// ---------------------------------------------------------------------------
__launch_bounds__(256)
__global__ void prep_kernel(const float* __restrict__ emb,
                            const float* __restrict__ Wc,
                            const float* __restrict__ Wq,
                            const float* __restrict__ Wk,
                            const float* __restrict__ Wv,
                            const float* __restrict__ Wo,
                            bf16* __restrict__ embbf,
                            bf16* __restrict__ Wts) {
  const size_t gid    = (size_t)blockIdx.x * 256 + threadIdx.x;
  const size_t stride = (size_t)gridDim.x * 256;

  for (size_t i = gid; i < (size_t)VOCAB_ * D_; i += stride)
    embbf[i] = (bf16)emb[i];

  const float* Ws[5] = {Wc, Wq, Wk, Wv, Wo};
  for (int w = 0; w < 5; ++w) {
    const float* W = Ws[w];
    bf16* img = Wts + (size_t)w * WT_IMG_ELEMS;
    for (size_t i = gid; i < 128 * 128; i += stride) {
      int k = (int)(i >> 7), n = (int)(i & 127);
      img[n * 136 + k] = (bf16)W[k * 128 + n];
    }
  }
}

// ---------------------------------------------------------------------------
// Kernel 1: per-(b,l) tree encoder.
//   e = emb[tokens] @ Wc + bc ; bottom-up subtree sums ; max over nodes ; relu
// One workgroup per tree; operands async-staged (bf16, preformatted) into
// LDS, 128x128x128 GEMM = 256 WMMAs, reduction in LDS, emits one enc row.
// ---------------------------------------------------------------------------
__launch_bounds__(256)
__global__ void tree_encode_kernel(const int*  __restrict__ tokens,
                                   const bf16* __restrict__ embbf,
                                   const bf16* __restrict__ WcTg,
                                   const float* __restrict__ bc,
                                   bf16*       __restrict__ enc) {
  __shared__ __align__(16) bf16  Abuf[128 * 136];  // gathered embeddings
  __shared__ __align__(16) bf16  WcT [128 * 136];  // Wc^T image
  __shared__ __align__(16) float hS  [128 * 132];  // e / tree accumulator
  __shared__ int   toks[128];
  __shared__ float red[2 * 128];

  const int tid  = threadIdx.x;
  const int bl   = blockIdx.x;          // flat (b,l), 0..4095
  const int lane = tid & 31;
  const int wave = tid >> 5;

  if (tid < NPT_) toks[tid] = tokens[(size_t)bl * NPT_ + tid];
  copy_flat_g2l(WcT, WcTg, WT_IMG_BYTES, tid);       // async weight stage
  __syncthreads();                                   // toks visible

  // Gather 127 embedding rows (256B each) -> Abuf rows (272B stride).
  for (int c = tid; c < NPT_ * 16; c += 256) {
    int row = c >> 4, seg = c & 15;
    copy16_g2l((char*)Abuf + row * 272 + seg * 16,
               (const char*)embbf + (size_t)toks[row] * 256 + seg * 16);
  }
  if (tid < 17) {                                    // zero pad row 127
    uint4 z; z.x = z.y = z.z = z.w = 0u;
    *(uint4*)((char*)Abuf + 127 * 272 + tid * 16) = z;
  }
  async_fence();
  __syncthreads();

  // 128x128x128 GEMM: wave = M-tile, 8 N-tiles, 4 K-steps => 32 WMMA/wave.
  v8f acc[8];
  #pragma unroll
  for (int i = 0; i < 8; ++i) acc[i] = zero_v8f();

  const int mtile = wave;
  #pragma unroll
  for (int ks = 0; ks < 4; ++ks) {
    v16bf a = load_frag_a(&Abuf[(mtile * 16) * 136 + ks * 32], 136, lane);
    #pragma unroll
    for (int nt = 0; nt < 8; ++nt) {
      v16bf bfr = load_frag_b(&WcT[(nt * 16) * 136 + ks * 32], 136, lane);
      acc[nt] = wmma_bf16(a, bfr, acc[nt]);
    }
  }

  // e + bc -> hS (f32)
  {
    const int n15  = lane & 15;
    const int moff = (lane >> 4) << 3;
    #pragma unroll
    for (int nt = 0; nt < 8; ++nt) {
      int n = nt * 16 + n15;
      float bcn = bc[n];
      #pragma unroll
      for (int j = 0; j < 8; ++j)
        hS[(mtile * 16 + moff + j) * 132 + n] = acc[nt][j] + bcn;
    }
  }
  __syncthreads();

  // Bottom-up subtree sums: h[p] += h[2p+1] + h[2p+2], levels 5..0.
  for (int level = 5; level >= 0; --level) {
    int start = (1 << level) - 1;
    int cnt   = 1 << level;
    for (int idx = tid; idx < cnt * D_; idx += 256) {
      int p = start + (idx >> 7);
      int d = idx & 127;
      hS[p * 132 + d] += hS[(2 * p + 1) * 132 + d] + hS[(2 * p + 2) * 132 + d];
    }
    __syncthreads();
  }

  // max over 127 nodes, then ReLU.
  {
    int d = tid & 127, half = tid >> 7;
    int n0 = half * 64, n1 = half ? NPT_ : 64;
    float m = -3.4e38f;
    for (int node = n0; node < n1; ++node) m = fmaxf(m, hS[node * 132 + d]);
    red[half * 128 + d] = m;
  }
  __syncthreads();
  if (tid < 128) {
    float v = fmaxf(red[tid], red[128 + tid]);
    enc[(size_t)bl * D_ + tid] = (bf16)fmaxf(v, 0.f);
  }
}

// ---------------------------------------------------------------------------
// Kernel 2: per-batch attention + pooling + classifier.
// ---------------------------------------------------------------------------
__device__ inline void gemm_tiles(const bf16* A, int lda,
                                  const bf16* Bt, int ldb,
                                  int mtile, int nbase, int ntiles, int ksteps,
                                  int lane, v8f* acc) {
  for (int i = 0; i < ntiles; ++i) acc[i] = zero_v8f();
  for (int ks = 0; ks < ksteps; ++ks) {
    v16bf a = load_frag_a(A + (mtile * 16) * lda + ks * 32, lda, lane);
    for (int i = 0; i < ntiles; ++i) {
      v16bf b = load_frag_b(Bt + ((nbase + i) * 16) * ldb + ks * 32, ldb, lane);
      acc[i] = wmma_bf16(a, b, acc[i]);
    }
  }
}

__launch_bounds__(256)
__global__ void attention_kernel(const bf16*  __restrict__ enc,
                                 const int*   __restrict__ mask,
                                 const bf16*  __restrict__ Wts,  // 5 images
                                 const float* __restrict__ Wl,
                                 const float* __restrict__ blv,
                                 float*       __restrict__ out) {
  // LDS map (bytes). PS aliases scoresS; outS aliases kS+scoresS (dead by
  // then); tmpS aliases qS.
  __shared__ __align__(16) char smem[124416];
  bf16*  encS    = (bf16*)(smem);            // [64][136]
  bf16*  qS      = (bf16*)(smem + 17408);    // [64][136]  (later: tmpS)
  bf16*  kS      = (bf16*)(smem + 34816);    // [64][136]
  float* scoresS = (float*)(smem + 52224);   // [64][68]
  bf16*  PS      = (bf16*)(smem + 52224);    // [64][136]  alias of scoresS
  bf16*  vT      = (bf16*)(smem + 69632);    // [128][72]  (v transposed)
  bf16*  WT      = (bf16*)(smem + 88064);    // [128][136] (current weight^T)
  float* outS    = (float*)(smem + 34816);   // [64][132]  alias of kS+scoresS
  float* redS    = (float*)(smem + 122880);  // 256 + 64 floats

  const int tid  = threadIdx.x;
  const int lane = tid & 31;
  const int wave = tid >> 5;
  const int b    = blockIdx.x;

  const int n15  = lane & 15;
  const int moff = (lane >> 4) << 3;

  const bf16* WqT = Wts + 1 * WT_IMG_ELEMS;
  const bf16* WkT = Wts + 2 * WT_IMG_ELEMS;
  const bf16* WvT = Wts + 3 * WT_IMG_ELEMS;
  const bf16* WoT = Wts + 4 * WT_IMG_ELEMS;

  // Stage enc rows (256B -> 272B stride) + Wq^T (flat image).
  for (int c = tid; c < 64 * 16; c += 256) {
    int r = c >> 4, seg = c & 15;
    copy16_g2l((char*)encS + r * 272 + seg * 16,
               (const char*)enc + (size_t)(b * 64 + r) * 256 + seg * 16);
  }
  copy_flat_g2l(WT, WqT, WT_IMG_BYTES, tid);
  async_fence();
  __syncthreads();

  v8f acc[4];
  const int pm = wave & 3;          // projection M-tile (64 rows -> 4 tiles)
  const int pn = (wave >> 2) * 4;   // projection N-tile base (8 -> 4 each)

  // q = enc @ Wq
  gemm_tiles(encS, 136, WT, 136, pm, pn, 4, 4, lane, acc);
  for (int i = 0; i < 4; ++i) {
    int n = (pn + i) * 16 + n15;
    #pragma unroll
    for (int j = 0; j < 8; ++j) qS[(pm * 16 + moff + j) * 136 + n] = (bf16)acc[i][j];
  }
  __syncthreads();
  copy_flat_g2l(WT, WkT, WT_IMG_BYTES, tid);
  async_fence();
  __syncthreads();

  // k = enc @ Wk  (row-major k serves directly as Bt for q @ k^T)
  gemm_tiles(encS, 136, WT, 136, pm, pn, 4, 4, lane, acc);
  for (int i = 0; i < 4; ++i) {
    int n = (pn + i) * 16 + n15;
    #pragma unroll
    for (int j = 0; j < 8; ++j) kS[(pm * 16 + moff + j) * 136 + n] = (bf16)acc[i][j];
  }
  __syncthreads();
  copy_flat_g2l(WT, WvT, WT_IMG_BYTES, tid);
  async_fence();
  __syncthreads();

  // v = enc @ Wv, stored transposed (vT[d][key]) -> Bt for P @ v.
  gemm_tiles(encS, 136, WT, 136, pm, pn, 4, 4, lane, acc);
  for (int i = 0; i < 4; ++i) {
    int n = (pn + i) * 16 + n15;
    #pragma unroll
    for (int j = 0; j < 8; ++j) vT[n * 72 + (pm * 16 + moff + j)] = (bf16)acc[i][j];
  }
  __syncthreads();

  // scores = (q @ k^T) / sqrt(D); 16 tiles over 8 waves.
  {
    const float scale = 0.088388347648318447f;  // 1/sqrt(128)
    int smt = wave >> 1;
    int snb = (wave & 1) * 2;
    gemm_tiles(qS, 136, kS, 136, smt, snb, 2, 4, lane, acc);
    for (int i = 0; i < 2; ++i) {
      int n = (snb + i) * 16 + n15;
      #pragma unroll
      for (int j = 0; j < 8; ++j)
        scoresS[(smt * 16 + moff + j) * 68 + n] = acc[i][j] * scale;
    }
  }
  copy_flat_g2l(WT, WoT, WT_IMG_BYTES, tid);  // Wv reads done pre-barrier
  async_fence();
  __syncthreads();

  // Masked softmax per row; P (bf16) overwrites the scores region.
  {
    int r = tid >> 2, qd = tid & 3, c0 = qd * 16;
    const int* mrow = mask + ((size_t)b * 64 + r) * 64;
    float vals[16];
    float mmax = -3.4e38f;
    #pragma unroll
    for (int i = 0; i < 16; ++i) {
      float s = scoresS[r * 68 + c0 + i];
      s = (mrow[c0 + i] > 0) ? s : -1.0e9f;
      vals[i] = s;
      mmax = fmaxf(mmax, s);
    }
    redS[tid] = mmax;
    __syncthreads();
    if (qd == 0) {
      redS[256 + r] = fmaxf(fmaxf(redS[r * 4], redS[r * 4 + 1]),
                            fmaxf(redS[r * 4 + 2], redS[r * 4 + 3]));
    }
    __syncthreads();
    float rowmax = redS[256 + r];
    float psum = 0.f;
    #pragma unroll
    for (int i = 0; i < 16; ++i) { vals[i] = __expf(vals[i] - rowmax); psum += vals[i]; }
    __syncthreads();
    redS[tid] = psum;
    __syncthreads();
    if (qd == 0) {
      redS[256 + r] = 1.f / (redS[r * 4] + redS[r * 4 + 1] +
                             redS[r * 4 + 2] + redS[r * 4 + 3]);
    }
    __syncthreads();
    float inv = redS[256 + r];
    #pragma unroll
    for (int i = 0; i < 16; ++i) PS[r * 136 + c0 + i] = (bf16)(vals[i] * inv);
  }
  __syncthreads();

  // tmp = P @ v (K=64) -> tmpS in the qS region (q dead).
  gemm_tiles(PS, 136, vT, 72, pm, pn, 4, 2, lane, acc);
  __syncthreads();          // all PS reads done before outS alias is written
  for (int i = 0; i < 4; ++i) {
    int n = (pn + i) * 16 + n15;
    #pragma unroll
    for (int j = 0; j < 8; ++j) qS[(pm * 16 + moff + j) * 136 + n] = (bf16)acc[i][j];
  }
  __syncthreads();

  // out = tmp @ Wo (f32 into outS alias region).
  gemm_tiles(qS, 136, WT, 136, pm, pn, 4, 4, lane, acc);
  for (int i = 0; i < 4; ++i) {
    int n = (pn + i) * 16 + n15;
    #pragma unroll
    for (int j = 0; j < 8; ++j) outS[(pm * 16 + moff + j) * 132 + n] = acc[i][j];
  }
  __syncthreads();

  // pooled = max over the 64 query rows.
  if (tid < 128) {
    float m = -3.4e38f;
    for (int r = 0; r < 64; ++r) m = fmaxf(m, outS[r * 132 + tid]);
    redS[tid] = m;
  }
  __syncthreads();

  // logits = pooled @ Wl + bl (128 x 30, scalar).
  if (tid < LABELS_) {
    float s = blv[tid];
    for (int d = 0; d < 128; ++d) s += redS[d] * Wl[d * LABELS_ + tid];
    out[b * LABELS_ + tid] = s;
  }
}

// ---------------------------------------------------------------------------
// Launch. Inputs (setup_inputs order):
//  0 tokens  1 mask  2 emb  3 Wc  4 bc  5 Wq  6 Wk  7 Wv  8 Wo  9 Wl  10 bl
// Output: f32 (B*LABELS).
// ---------------------------------------------------------------------------
extern "C" void kernel_launch(void* const* d_in, const int* in_sizes, int n_in,
                              void* d_out, int out_size, void* d_ws, size_t ws_size,
                              hipStream_t stream) {
  (void)in_sizes; (void)n_in; (void)out_size; (void)ws_size;

  const int*   tokens = (const int*)  d_in[0];
  const int*   mask   = (const int*)  d_in[1];
  const float* emb    = (const float*)d_in[2];
  const float* Wc     = (const float*)d_in[3];
  const float* bc     = (const float*)d_in[4];
  const float* Wq     = (const float*)d_in[5];
  const float* Wk     = (const float*)d_in[6];
  const float* Wv     = (const float*)d_in[7];
  const float* Wo     = (const float*)d_in[8];
  const float* Wl     = (const float*)d_in[9];
  const float* bl     = (const float*)d_in[10];
  float* out = (float*)d_out;

  char* ws = (char*)d_ws;
  bf16* enc   = (bf16*)(ws + WS_ENC_OFF);
  bf16* Wts   = (bf16*)(ws + WS_WT_OFF);
  bf16* embbf = (bf16*)(ws + WS_EMB_OFF);

  prep_kernel<<<480, 256, 0, stream>>>(emb, Wc, Wq, Wk, Wv, Wo, embbf, Wts);
  tree_encode_kernel<<<B_ * L_, 256, 0, stream>>>(tokens, embbf,
                                                  Wts /* image 0 = Wc^T */,
                                                  bc, enc);
  attention_kernel<<<B_, 256, 0, stream>>>(enc, mask, Wts, Wl, bl, out);
}
